// PointTransformerMat_47459388620946
// MI455X (gfx1250) — matrix operand order repaired and checked by reading, not compiled, not run on hardware
//
#include <hip/hip_runtime.h>
#include <hip/hip_bf16.h>
#include <stdint.h>

// Problem constants (from reference)
#define Cc   256        // channels
#define Hh   4          // heads
#define Ll   4          // layers
#define NS   32         // group size
#define Bb   4          // batch
#define Nn   16384      // points
#define NP   512        // npoint = round_up_pow2(N/NS)
#define DH   64         // head dim
#define QKVW (3*Cc)     // 768

// ---------------- WMMA bf16 plumbing (CDNA5 16x16x32) ----------------
typedef __attribute__((ext_vector_type(16))) __bf16 v16bf;
typedef __attribute__((ext_vector_type(8)))  float  v8f;

union BF16x16 { v16bf v; unsigned short u[16]; uint4 q[2]; };

__device__ __forceinline__ unsigned short f2bf(float x) {
  union { float f; unsigned u; } c; c.f = x;
  unsigned r = c.u + 0x7FFFu + ((c.u >> 16) & 1u);   // round-to-nearest-even
  return (unsigned short)(r >> 16);
}

__device__ __forceinline__ v8f wmma_bf16(v16bf a, v16bf b, v8f c) {
  // D = A(16x32) * B(32x16) + C, f32 accum
  return __builtin_amdgcn_wmma_f32_16x16x32_bf16(false, a, false, b, (short)0, c, false, false);
}

// A fragment: rows m0..+15, cols k0..+31 of row-major bf16 matrix (lda elems).
// Lane layout: lanes 0-15 M=lane K-base+0; lanes 16-31 M=lane-16 K-base+16;
// per lane the 16 elems are one contiguous run [kb..kb+15] -> 2x b128.
__device__ __forceinline__ v16bf load_a_frag(const unsigned short* a, int lda, int m0, int k0) {
  int lane = threadIdx.x & 31;
  const unsigned short* p = a + (size_t)(m0 + (lane & 15)) * lda + k0 + ((lane >> 4) << 4);
  BF16x16 f;
  f.q[0] = *(const uint4*)(p);
  f.q[1] = *(const uint4*)(p + 8);
  return f.v;
}

// B fragment from COLUMN-major storage bt[N][K] (i.e. W^T, or "B^T of row-major"):
// lane col = n0+(lane&15), K run = k0 + 16*(lane>>4) .. +15 -> contiguous, 2x b128.
__device__ __forceinline__ v16bf load_bT_frag(const unsigned short* bt, int ldk, int k0, int n0) {
  int lane = threadIdx.x & 31;
  const unsigned short* p = bt + (size_t)(n0 + (lane & 15)) * ldk + k0 + ((lane >> 4) << 4);
  BF16x16 f;
  f.q[0] = *(const uint4*)(p);
  f.q[1] = *(const uint4*)(p + 8);
  return f.v;
}

// ---------------- misc small kernels ----------------
// fp32 W[L][K][N] -> bf16 W^T[L][N][K] (per-layer transpose at conversion time)
__global__ void cvt_t_kernel(const float* __restrict__ s, unsigned short* __restrict__ d,
                             int K, int N, int L) {
  long tot = (long)L * K * N;
  long i = (long)blockIdx.x * blockDim.x + threadIdx.x;
  long st = (long)gridDim.x * blockDim.x;
  for (; i < tot; i += st) {
    int l = (int)(i / ((long)K * N));
    int rem = (int)(i - (long)l * K * N);
    int k = rem / N, n = rem % N;
    d[(size_t)l * K * N + (size_t)n * K + k] = f2bf(s[i]);
  }
}

__global__ void zero_kernel(float* p, long n) {
  long i = (long)blockIdx.x * blockDim.x + threadIdx.x;
  long st = (long)gridDim.x * blockDim.x;
  for (; i < n; i += st) p[i] = 0.f;
}

// ---------------- FPS: 1 block per batch, dist kept in registers ----------------
__global__ void __launch_bounds__(256) fps_kernel(const float* __restrict__ xyz, int* __restrict__ fps_idx) {
  int b = blockIdx.x, t = threadIdx.x;
  const float* base = xyz + (size_t)b * Nn * 3;
  float dist[64];
#pragma unroll
  for (int j = 0; j < 64; ++j) dist[j] = 3.4e38f;
  __shared__ float smax[256];
  __shared__ int   simax[256];
  __shared__ int   sfar;
  if (t == 0) sfar = 0;
  __syncthreads();
  for (int i = 0; i < NP; ++i) {
    int far = sfar;
    if (t == 0) fps_idx[b * NP + i] = far;
    float px = base[far*3+0], py = base[far*3+1], pz = base[far*3+2];
    float lmax = -1.f; int lidx = 0;
#pragma unroll
    for (int j = 0; j < 64; ++j) {
      int n = t + j * 256;
      float dx = base[n*3+0]-px, dy = base[n*3+1]-py, dz = base[n*3+2]-pz;
      float d = dx*dx + dy*dy + dz*dz;
      float nd = d < dist[j] ? d : dist[j];
      dist[j] = nd;
      if (nd > lmax) { lmax = nd; lidx = n; }
    }
    smax[t] = lmax; simax[t] = lidx;
    __syncthreads();
    for (int s = 128; s > 0; s >>= 1) {
      if (t < s) {
        float ov = smax[t+s]; int oi = simax[t+s];
        if (ov > smax[t] || (ov == smax[t] && oi < simax[t])) { smax[t] = ov; simax[t] = oi; }
      }
      __syncthreads();
    }
    if (t == 0) sfar = simax[0];
    __syncthreads();
  }
}

// ---------------- KNN top-32 + centroid, 1 block per (b,m) ----------------
__global__ void __launch_bounds__(256) knn_kernel(const float* __restrict__ xyz,
                                                  const float* __restrict__ pd,
                                                  const int* __restrict__ fps_idx,
                                                  int* __restrict__ knn_idx,
                                                  float* __restrict__ centroids) {
  int g = blockIdx.x, b = g >> 9, t = threadIdx.x;
  const float* base = xyz + (size_t)b * Nn * 3;
  int fi = fps_idx[g];
  float px = base[fi*3+0], py = base[fi*3+1], pz = base[fi*3+2];
  float dist[64];
#pragma unroll
  for (int j = 0; j < 64; ++j) {
    int n = t + j * 256;
    float dx = base[n*3+0]-px, dy = base[n*3+1]-py, dz = base[n*3+2]-pz;
    dist[j] = dx*dx + dy*dy + dz*dz;
  }
  __shared__ float smin[256];
  __shared__ int   simin[256];
  __shared__ int   sel[NS];
  for (int r = 0; r < NS; ++r) {
    float lmin = 3.4e38f; int lidx = 0;
#pragma unroll
    for (int j = 0; j < 64; ++j)
      if (dist[j] < lmin) { lmin = dist[j]; lidx = t + j * 256; }
    smin[t] = lmin; simin[t] = lidx;
    __syncthreads();
    for (int s = 128; s > 0; s >>= 1) {
      if (t < s) {
        if (smin[t+s] < smin[t] || (smin[t+s] == smin[t] && simin[t+s] < simin[t])) {
          smin[t] = smin[t+s]; simin[t] = simin[t+s];
        }
      }
      __syncthreads();
    }
    int win = simin[0];
    if (t == 0) { knn_idx[g * NS + r] = win; sel[r] = win; }
    if ((win & 255) == t) {
#pragma unroll
      for (int j = 0; j < 64; ++j) if (t + j * 256 == win) dist[j] = 3.4e38f;
    }
    __syncthreads();
  }
  __shared__ float cacc[NS][3];
  if (t < NS) {
    const float* p = pd + ((size_t)b * Nn + sel[t]) * Cc;
    cacc[t][0] = p[0]; cacc[t][1] = p[1]; cacc[t][2] = p[2];
  }
  __syncthreads();
  if (t < 3) {
    float s = 0.f;
    for (int r = 0; r < NS; ++r) s += cacc[r][t];
    centroids[(size_t)g * 3 + t] = s * (1.f / NS);
  }
}

// ---------------- centroid assignment (argmin over 512) ----------------
__global__ void __launch_bounds__(256) cc_kernel(const float* __restrict__ xyz,
                                                 const float* __restrict__ centroids,
                                                 int* __restrict__ cc_idx) {
  __shared__ float cen[NP][3];
  int t = threadIdx.x;
  int p = blockIdx.x * 256 + t;
  int b = p >> 14, n = p & (Nn - 1);
  for (int i = t; i < NP * 3; i += 256) cen[i / 3][i % 3] = centroids[(size_t)b * NP * 3 + i];
  __syncthreads();
  const float* q = xyz + ((size_t)b * Nn + n) * 3;
  float px = q[0], py = q[1], pz = q[2];
  float best = 3.4e38f; int bi = 0;
  for (int m = 0; m < NP; ++m) {
    float dx = px - cen[m][0], dy = py - cen[m][1], dz = pz - cen[m][2];
    float d = dx*dx + dy*dy + dz*dz;
    if (d < best) { best = d; bi = m; }
  }
  cc_idx[p] = bi;
}

// ---------------- fused local stacked attention ----------------
__device__ void ln32(float (*xf)[Cc], const float* gg, const float* bb,
                     unsigned short (*hb)[Cc], bool writeback, float* gout) {
  int t = threadIdx.x, wave = t >> 5, lane = t & 31;
  for (int rr = 0; rr < 4; ++rr) {
    int row = wave * 4 + rr;
    float s = 0.f;
    for (int c = lane; c < Cc; c += 32) s += xf[row][c];
#pragma unroll
    for (int o = 16; o; o >>= 1) s += __shfl_xor(s, o, 32);
    float mu = s * (1.f / Cc);
    float v = 0.f;
    for (int c = lane; c < Cc; c += 32) { float d = xf[row][c] - mu; v += d * d; }
#pragma unroll
    for (int o = 16; o; o >>= 1) v += __shfl_xor(v, o, 32);
    float rs = rsqrtf(v * (1.f / Cc) + 1e-5f);
    for (int c = lane; c < Cc; c += 32) {
      float val = (xf[row][c] - mu) * rs * gg[c] + bb[c];
      if (hb) hb[row][c] = f2bf(val);
      if (writeback) xf[row][c] = val;
      if (gout) gout[row * Cc + c] = val;
    }
  }
}

// C(32xN) = A(32x256,bf16 row-major) * W^T(NxK bf16 col-major weights) + bias.
// mode 0: write bf16 to outb (ld=N), optionally mirroring V third transposed into vT[256][NS];
// mode 1: accumulate fp32 into outf (residual).
__device__ void gemm32(const unsigned short* A, int lda,
                       const unsigned short* Wt, int N, const float* bias,
                       unsigned short* outb, unsigned short* vT,
                       float (*outf)[Cc], int mode) {
  int t = threadIdx.x, wave = t >> 5, lane = t & 31;
  for (int nt = wave; nt < (N >> 4); nt += 8) {
    int n0 = nt << 4;
    v8f acc0 = {}, acc1 = {};
#pragma unroll
    for (int k0 = 0; k0 < Cc; k0 += 32) {
      __builtin_prefetch(Wt + (size_t)(n0 + (lane & 15)) * Cc + k0 + 32, 0, 0);
      v16bf bf = load_bT_frag(Wt, Cc, k0, n0);
      v16bf a0 = load_a_frag(A, lda, 0, k0);
      v16bf a1 = load_a_frag(A, lda, 16, k0);
      acc0 = wmma_bf16(a0, bf, acc0);
      acc1 = wmma_bf16(a1, bf, acc1);
    }
    int col = n0 + (lane & 15);
    float bv = bias[col];
    int rhal = (lane >> 4) << 3;
#pragma unroll
    for (int r = 0; r < 8; ++r) {
      float v0 = acc0[r] + bv, v1 = acc1[r] + bv;
      int r0 = rhal + r, r1 = 16 + rhal + r;
      if (mode) {
        outf[r0][col] += v0; outf[r1][col] += v1;
      } else {
        unsigned short h0 = f2bf(v0), h1 = f2bf(v1);
        outb[r0 * N + col] = h0; outb[r1 * N + col] = h1;
        if (vT && col >= 2 * Cc) {           // mirror V transposed: vT[d_all][token]
          int cv = col - 2 * Cc;
          vT[cv * NS + r0] = h0; vT[cv * NS + r1] = h1;
        }
      }
    }
  }
}

__global__ void __launch_bounds__(256) local_attn_kernel(
    const float* __restrict__ pd, const int* __restrict__ knn_idx,
    const unsigned short* __restrict__ wqkv, const float* __restrict__ bqkv,
    const unsigned short* __restrict__ wo,   const float* __restrict__ bo,
    const unsigned short* __restrict__ ffw,  const float* __restrict__ ffb,
    const float* __restrict__ ln1g, const float* __restrict__ ln1b,
    const float* __restrict__ ln2g, const float* __restrict__ ln2b,
    const float* __restrict__ fing, const float* __restrict__ finb,
    float* __restrict__ loc_out, float* __restrict__ tokens) {
  __shared__ __align__(16) float          xf[NS][Cc];        // 32 KB fp32 residual
  __shared__ __align__(16) unsigned short hb[NS][Cc];        // 16 KB LN out bf16
  __shared__ __align__(16) unsigned short qkvb[NS][QKVW];    // 48 KB q|k|v bf16
  __shared__ __align__(16) unsigned short vTl[Cc][NS];       // 16 KB V^T (4 heads x 64 x 32)
  __shared__ __align__(16) unsigned short ob[NS][Cc];        // 16 KB attn out bf16
  __shared__ __align__(16) float          sc[Hh][NS][NS];    // 16 KB scores fp32
  __shared__ __align__(16) unsigned short scb[Hh][NS][NS];   //  8 KB probs bf16
  __shared__ int gidx[NS];

  int g = blockIdx.x, b = g >> 9;
  int t = threadIdx.x, wave = t >> 5, lane = t & 31;

  if (t < NS) gidx[t] = knn_idx[g * NS + t];
  __syncthreads();
  for (int rc = t; rc < NS * (Cc / 4); rc += 256) {          // float4 gather
    int r = rc >> 6, c4 = rc & 63;
    const float4* src = (const float4*)(pd + ((size_t)b * Nn + gidx[r]) * Cc);
    ((float4*)&xf[r][0])[c4] = src[c4];
  }
  __syncthreads();

  for (int l = 0; l < Ll; ++l) {
    const unsigned short* Wq = wqkv + (size_t)l * Cc * QKVW;  // [768][256] col-major weights
    const unsigned short* Wp = wo   + (size_t)l * Cc * Cc;    // [256][256]
    const unsigned short* Wf = ffw  + (size_t)l * Cc * Cc;

    ln32(xf, ln1g + l * Cc, ln1b + l * Cc, hb, false, nullptr);
    __syncthreads();
    gemm32(&hb[0][0], Cc, Wq, QKVW, bqkv + l * QKVW, &qkvb[0][0], &vTl[0][0], nullptr, 0);
    __syncthreads();

    { // attention: wave -> (head, m-tile)
      int h = wave >> 1, mt = wave & 1, m0 = mt * 16;
      const unsigned short* qp = &qkvb[0][0] + h * DH;
      const unsigned short* kp = &qkvb[0][0] + Cc + h * DH;   // rows=tokens, contiguous K
#pragma unroll
      for (int nt = 0; nt < 2; ++nt) {
        v8f acc = {};
#pragma unroll
        for (int k0 = 0; k0 < DH; k0 += 32) {
          v16bf af = load_a_frag(qp, QKVW, m0, k0);
          v16bf bf = load_bT_frag(kp, QKVW, k0, nt * 16);     // K^T, contiguous
          acc = wmma_bf16(af, bf, acc);
        }
        int col = nt * 16 + (lane & 15);
        int rb = m0 + ((lane >> 4) << 3);
#pragma unroll
        for (int r = 0; r < 8; ++r) sc[h][rb + r][col] = acc[r] * 0.125f; // 1/sqrt(64)
      }
      __syncthreads();
      if (lane < 16) { // one row per lane
        int row = m0 + lane;
        float mx = -3.4e38f, e[NS], s = 0.f;
#pragma unroll
        for (int j = 0; j < NS; ++j) mx = fmaxf(mx, sc[h][row][j]);
#pragma unroll
        for (int j = 0; j < NS; ++j) { e[j] = __expf(sc[h][row][j] - mx); s += e[j]; }
        float inv = 1.f / s;
#pragma unroll
        for (int j = 0; j < NS; ++j) scb[h][row][j] = f2bf(e[j] * inv);
      }
      __syncthreads();
#pragma unroll
      for (int nt = 0; nt < 4; ++nt) { // o = P @ V via V^T tile, K=32
        v8f acc = {};
        v16bf af = load_a_frag(&scb[h][0][0], NS, m0, 0);
        v16bf bf = load_bT_frag(&vTl[h * DH][0], NS, 0, nt * 16);
        acc = wmma_bf16(af, bf, acc);
        int col = nt * 16 + (lane & 15);
        int rb = m0 + ((lane >> 4) << 3);
#pragma unroll
        for (int r = 0; r < 8; ++r) ob[rb + r][h * DH + col] = f2bf(acc[r]);
      }
    }
    __syncthreads();
    gemm32(&ob[0][0], Cc, Wp, Cc, bo + l * Cc, nullptr, nullptr, xf, 1);   // x += o@Wo + bo
    __syncthreads();
    ln32(xf, ln2g + l * Cc, ln2b + l * Cc, hb, false, nullptr);
    __syncthreads();
    gemm32(&hb[0][0], Cc, Wf, Cc, ffb + l * Cc, nullptr, nullptr, xf, 1);  // x += h2@Wf + fb
    __syncthreads();
  }

  ln32(xf, fing, finb, nullptr, true, loc_out + (size_t)g * NS * Cc); // final LN
  __syncthreads();
  { // group token = mean over 32 rows
    float s = 0.f;
    for (int r = 0; r < NS; ++r) s += xf[r][t];
    tokens[(size_t)g * Cc + t] = s * (1.f / NS);
  }
}

// ---------------- scatter-mean into d_out ----------------
__global__ void __launch_bounds__(256) scatter_kernel(const float* __restrict__ loc,
                                                      const int* __restrict__ knn_idx,
                                                      float* __restrict__ outp,
                                                      float* __restrict__ cnt) {
  int row = blockIdx.x, c = threadIdx.x;
  int b = row >> 14;                      // NP*NS = 16384 rows / batch
  int fi = b * Nn + knn_idx[row];
  atomicAdd(&outp[(size_t)fi * Cc + c], loc[(size_t)row * Cc + c]);
  if (c == 0) atomicAdd(&cnt[fi], 1.f);
}

// ---------------- global attention pieces ----------------
__global__ void __launch_bounds__(256) ln_glb_kernel(const float* __restrict__ x,
                                                     const float* __restrict__ gg,
                                                     const float* __restrict__ bb,
                                                     unsigned short* __restrict__ hb,
                                                     float* __restrict__ fout, int M) {
  int row = blockIdx.x * 8 + (threadIdx.x >> 5);
  int lane = threadIdx.x & 31;
  if (row >= M) return;
  const float* xr = x + (size_t)row * Cc;
  float s = 0.f;
  for (int c = lane; c < Cc; c += 32) s += xr[c];
#pragma unroll
  for (int o = 16; o; o >>= 1) s += __shfl_xor(s, o, 32);
  float mu = s * (1.f / Cc);
  float v = 0.f;
  for (int c = lane; c < Cc; c += 32) { float d = xr[c] - mu; v += d * d; }
#pragma unroll
  for (int o = 16; o; o >>= 1) v += __shfl_xor(v, o, 32);
  float rs = rsqrtf(v * (1.f / Cc) + 1e-5f);
  for (int c = lane; c < Cc; c += 32) {
    float val = (xr[c] - mu) * rs * gg[c] + bb[c];
    if (hb)   hb[(size_t)row * Cc + c] = f2bf(val);
    if (fout) fout[(size_t)row * Cc + c] = val;
  }
}

// one 16x16 output tile per wave (m-tile fastest); K=256, col-major weights.
// Optionally mirrors the V third into vT[b][h][64][512] (token-contiguous).
__global__ void __launch_bounds__(256) gemm_glb_kernel(const unsigned short* __restrict__ A, int M,
                                                       const unsigned short* __restrict__ Wt, int N,
                                                       const float* __restrict__ bias,
                                                       unsigned short* __restrict__ outb,
                                                       unsigned short* __restrict__ vT,
                                                       float* __restrict__ outf, int mode) {
  int wave = threadIdx.x >> 5, lane = threadIdx.x & 31;
  int tid = blockIdx.x * 8 + wave;
  int mtiles = M >> 4;
  int mt = tid % mtiles, nt = tid / mtiles;
  if (nt >= (N >> 4)) return;   // wave-uniform: EXEC stays all-ones for WMMA
  int m0 = mt << 4, n0 = nt << 4;
  v8f acc = {};
#pragma unroll
  for (int k0 = 0; k0 < Cc; k0 += 32) {
    __builtin_prefetch(Wt + (size_t)(n0 + (lane & 15)) * Cc + k0 + 32, 0, 0);
    v16bf af = load_a_frag(A, Cc, m0, k0);
    v16bf bf = load_bT_frag(Wt, Cc, k0, n0);
    acc = wmma_bf16(af, bf, acc);
  }
  int col = n0 + (lane & 15);
  float bv = bias[col];
  int rb = m0 + ((lane >> 4) << 3);
#pragma unroll
  for (int r = 0; r < 8; ++r) {
    float val = acc[r] + bv;
    int row = rb + r;
    size_t o = (size_t)row * N + col;
    if (mode) outf[o] += val;
    else {
      unsigned short hv = f2bf(val);
      outb[o] = hv;
      if (vT && col >= 2 * Cc) {
        int cv = col - 2 * Cc;                 // 0..255 across heads
        int bidx = row >> 9, s = row & (NP - 1);
        vT[(((size_t)(bidx * Hh) + (cv >> 6)) * DH + (cv & 63)) * NP + s] = hv;
      }
    }
  }
}

// one (batch, head, 32-row q-tile) per block; 32x512 score tile in LDS
__global__ void __launch_bounds__(256) glb_attn_kernel(const unsigned short* __restrict__ qkv,
                                                       const unsigned short* __restrict__ vT,
                                                       unsigned short* __restrict__ oout) {
  __shared__ __align__(16) float          sc[NS][NP];   // 64 KB
  __shared__ __align__(16) unsigned short scb[NS][NP];  // 32 KB
  int blk = blockIdx.x;
  int qt = blk & 15, h = (blk >> 4) & 3, b = blk >> 6;
  int wave = threadIdx.x >> 5, lane = threadIdx.x & 31;
  const unsigned short* qbase = qkv + ((size_t)(b * NP + qt * 32)) * QKVW + h * DH;
  const unsigned short* kbase = qkv + ((size_t)b * NP) * QKVW + Cc + h * DH;
  const unsigned short* vtb   = vT + ((size_t)(b * Hh + h) * DH) * NP;

  for (int t2 = wave; t2 < 64; t2 += 8) {             // 2 m-tiles x 32 n-tiles
    int mt = t2 & 1, nt = t2 >> 1;
    int m0 = mt * 16, n0 = nt * 16;
    v8f acc = {};
#pragma unroll
    for (int k0 = 0; k0 < DH; k0 += 32) {
      v16bf af = load_a_frag(qbase, QKVW, m0, k0);
      v16bf bf = load_bT_frag(kbase, QKVW, k0, n0);   // K^T, contiguous
      acc = wmma_bf16(af, bf, acc);
    }
    int col = n0 + (lane & 15);
    int rb = m0 + ((lane >> 4) << 3);
#pragma unroll
    for (int r = 0; r < 8; ++r) sc[rb + r][col] = acc[r] * 0.125f;
  }
  __syncthreads();
  { // softmax over 512 keys; 8-lane subgroup per row
    int row = wave * 4 + (lane >> 3);
    int li = lane & 7;
    float mx = -3.4e38f;
    for (int j = li; j < NP; j += 8) mx = fmaxf(mx, sc[row][j]);
#pragma unroll
    for (int o = 4; o; o >>= 1) mx = fmaxf(mx, __shfl_xor(mx, o, 32));
    float s = 0.f;
    for (int j = li; j < NP; j += 8) { float e = __expf(sc[row][j] - mx); sc[row][j] = e; s += e; }
#pragma unroll
    for (int o = 4; o; o >>= 1) s += __shfl_xor(s, o, 32);
    float inv = 1.f / s;
    for (int j = li; j < NP; j += 8) scb[row][j] = f2bf(sc[row][j] * inv);
  }
  __syncthreads();
  { // o = P(32x512) @ V(512x64) via V^T: 8 tiles, one per wave
    int mt = wave & 1, nt = wave >> 1;
    int m0 = mt * 16, n0 = nt * 16;
    v8f acc = {};
#pragma unroll 4
    for (int k0 = 0; k0 < NP; k0 += 32) {
      v16bf af = load_a_frag(&scb[0][0], NP, m0, k0);
      v16bf bf = load_bT_frag(vtb, NP, k0, n0);       // contiguous tokens
      acc = wmma_bf16(af, bf, acc);
    }
    int col = n0 + (lane & 15);
    int rb = m0 + ((lane >> 4) << 3);
#pragma unroll
    for (int r = 0; r < 8; ++r)
      oout[((size_t)(b * NP + qt * 32 + rb + r)) * Cc + h * DH + col] = f2bf(acc[r]);
  }
}

// ---------------- finalize: out = recon/cnt + glb[cc_idx] ----------------
__global__ void __launch_bounds__(256) finalize_kernel(const float* __restrict__ glb,
                                                       const int* __restrict__ cc_idx,
                                                       const float* __restrict__ cnt,
                                                       float* __restrict__ outp) {
  int p = blockIdx.x, c = threadIdx.x;
  int b = p >> 14;
  float cv = fmaxf(cnt[p], 1.f);
  int gi = b * NP + cc_idx[p];
  size_t o = (size_t)p * Cc + c;
  outp[o] = outp[o] / cv + glb[(size_t)gi * Cc + c];
}

// ---------------- host orchestration ----------------
extern "C" void kernel_launch(void* const* d_in, const int* in_sizes, int n_in,
                              void* d_out, int out_size, void* d_ws, size_t ws_size,
                              hipStream_t stream) {
  (void)in_sizes; (void)n_in; (void)out_size; (void)ws_size;
  const float* xyz = (const float*)d_in[0];
  const float* pd  = (const float*)d_in[1];
  // param leaves assumed in dict insertion order: ln1_g, ln1_b, wqkv, bqkv, wo, bo,
  // ln2_g, ln2_b, ffw, ffb, fin_g, fin_b  (loc then glb)
  enum { LN1G = 0, LN1B, WQKV, BQKV, WO, BO, LN2G, LN2B, FFW, FFB, FING, FINB };
  const float* P[24];
  for (int i = 0; i < 24; ++i) P[i] = (const float*)d_in[2 + i];
  const float* const* G = P + 12;

  char* ws = (char*)d_ws;
  size_t off = 0;
  auto take = [&](size_t bytes) { char* p = ws + off; off += (bytes + 255) & ~(size_t)255; return p; };

  const size_t SZ_WQKV = (size_t)Ll * Cc * QKVW;   // 786432
  const size_t SZ_WSQ  = (size_t)Ll * Cc * Cc;     // 262144

  unsigned short* wqkvL = (unsigned short*)take(SZ_WQKV * 2);
  unsigned short* woL   = (unsigned short*)take(SZ_WSQ * 2);
  unsigned short* ffwL  = (unsigned short*)take(SZ_WSQ * 2);
  unsigned short* wqkvG = (unsigned short*)take(SZ_WQKV * 2);
  unsigned short* woG   = (unsigned short*)take(SZ_WSQ * 2);
  unsigned short* ffwG  = (unsigned short*)take(SZ_WSQ * 2);
  int*   fpsI = (int*)take((size_t)Bb * NP * 4);
  int*   knnI = (int*)take((size_t)Bb * NP * NS * 4);
  float* cent = (float*)take((size_t)Bb * NP * 3 * 4);
  int*   ccI  = (int*)take((size_t)Bb * Nn * 4);
  float* cnt  = (float*)take((size_t)Bb * Nn * 4);
  float* loc  = (float*)take((size_t)Bb * NP * NS * Cc * 4);
  float* xg   = (float*)take((size_t)Bb * NP * Cc * 4);
  unsigned short* hg  = (unsigned short*)take((size_t)Bb * NP * Cc * 2);
  unsigned short* qkg = (unsigned short*)take((size_t)Bb * NP * QKVW * 2);
  unsigned short* vtg = (unsigned short*)take((size_t)Bb * Hh * DH * NP * 2);
  unsigned short* og  = (unsigned short*)take((size_t)Bb * NP * Cc * 2);

  float* outp = (float*)d_out;

  // weights -> bf16, transposed per layer to column-major (L2-resident, ~5 MB)
  cvt_t_kernel<<<1024, 256, 0, stream>>>(P[WQKV], wqkvL, Cc, QKVW, Ll);
  cvt_t_kernel<<<512,  256, 0, stream>>>(P[WO],   woL,   Cc, Cc,   Ll);
  cvt_t_kernel<<<512,  256, 0, stream>>>(P[FFW],  ffwL,  Cc, Cc,   Ll);
  cvt_t_kernel<<<1024, 256, 0, stream>>>(G[WQKV], wqkvG, Cc, QKVW, Ll);
  cvt_t_kernel<<<512,  256, 0, stream>>>(G[WO],   woG,   Cc, Cc,   Ll);
  cvt_t_kernel<<<512,  256, 0, stream>>>(G[FFW],  ffwG,  Cc, Cc,   Ll);

  zero_kernel<<<4096, 256, 0, stream>>>(outp, (long)Bb * Nn * Cc);
  zero_kernel<<<64,   256, 0, stream>>>(cnt,  (long)Bb * Nn);

  fps_kernel<<<Bb, 256, 0, stream>>>(xyz, fpsI);
  knn_kernel<<<Bb * NP, 256, 0, stream>>>(xyz, pd, fpsI, knnI, cent);
  cc_kernel<<<Bb * Nn / 256, 256, 0, stream>>>(xyz, cent, ccI);

  local_attn_kernel<<<Bb * NP, 256, 0, stream>>>(
      pd, knnI, wqkvL, P[BQKV], woL, P[BO], ffwL, P[FFB],
      P[LN1G], P[LN1B], P[LN2G], P[LN2B], P[FING], P[FINB], loc, xg);

  scatter_kernel<<<Bb * NP * NS, 256, 0, stream>>>(loc, knnI, outp, cnt);

  const int Mg = Bb * NP;                 // 2048 rows
  for (int l = 0; l < Ll; ++l) {
    ln_glb_kernel<<<Mg / 8, 256, 0, stream>>>(xg, G[LN1G] + l * Cc, G[LN1B] + l * Cc, hg, (float*)nullptr, Mg);
    gemm_glb_kernel<<<(Mg / 16) * (QKVW / 16) / 8, 256, 0, stream>>>(
        hg, Mg, wqkvG + (size_t)l * Cc * QKVW, QKVW, G[BQKV] + l * QKVW, qkg, vtg, (float*)nullptr, 0);
    glb_attn_kernel<<<Bb * Hh * 16, 256, 0, stream>>>(qkg, vtg, og);
    gemm_glb_kernel<<<(Mg / 16) * (Cc / 16) / 8, 256, 0, stream>>>(
        og, Mg, woG + (size_t)l * Cc * Cc, Cc, G[BO] + l * Cc, (unsigned short*)nullptr,
        (unsigned short*)nullptr, xg, 1);
    ln_glb_kernel<<<Mg / 8, 256, 0, stream>>>(xg, G[LN2G] + l * Cc, G[LN2B] + l * Cc, hg, (float*)nullptr, Mg);
    gemm_glb_kernel<<<(Mg / 16) * (Cc / 16) / 8, 256, 0, stream>>>(
        hg, Mg, ffwG + (size_t)l * Cc * Cc, Cc, G[FFB] + l * Cc, (unsigned short*)nullptr,
        (unsigned short*)nullptr, xg, 1);
  }
  ln_glb_kernel<<<Mg / 8, 256, 0, stream>>>(xg, G[FING], G[FINB], (unsigned short*)nullptr, xg, Mg);

  finalize_kernel<<<Bb * Nn, 256, 0, stream>>>(xg, ccI, cnt, outp);
}
